// ST_DMLM_61177514164479
// MI455X (gfx1250) — matrix operand order repaired and checked by reading, not compiled, not run on hardware
//
#include <hip/hip_runtime.h>
#include <hip/hip_bf16.h>

// ---------------------------------------------------------------------------
// ST-DMLM on MI455X (gfx1250): batch-folded GCN GEMMs via v_wmma_f32_16x16x32_f16,
// LDS staged with GLOBAL_LOAD_ASYNC_TO_LDS_B128 (ASYNCcnt) and the Tensor Data
// Mover (TENSOR_LOAD_TO_LDS / TENSORcnt) for weight tiles, double-buffered.
// Dims: B=64 T=12 N=1024 C=1 K=3 H=64 M=32 HOR=12 OUT=1 DIN=65 G4=256
// ---------------------------------------------------------------------------

typedef __attribute__((ext_vector_type(16))) _Float16 v16h;
typedef __attribute__((ext_vector_type(8)))  _Float16 h8;
typedef __attribute__((ext_vector_type(8)))  float    v8f;
typedef __attribute__((ext_vector_type(4)))  unsigned u32x4;
typedef __attribute__((ext_vector_type(8)))  unsigned u32x8;

#define NB   64
#define NT   12
#define NN   1024
#define NH   64
#define NDIN 65
#define NG4  256
#define NCOL (NB * NDIN)   /* 4160 GEMM1 columns            */
#define PPAD 224           /* 195 (=K*DIN) padded to 7*32   */

// ---- CDNA5 async / TDM helpers --------------------------------------------
__device__ __forceinline__ unsigned lds_lo32(const void* p) {
    // flat pointers into LDS: low 32 bits are the LDS byte address (ISA 10.2)
    return (unsigned)(unsigned long long)p;
}
__device__ __forceinline__ void async_ld16(unsigned lds_addr, const void* g) {
    asm volatile("global_load_async_to_lds_b128 %0, %1, off"
                 :: "v"(lds_addr), "v"(g) : "memory");
}
__device__ __forceinline__ void wait_async0() {
    asm volatile("s_wait_asynccnt 0x0" ::: "memory");
}
// 2-D tile DMA: tile_h rows of tile_w elements (2B), row stride = stride elems.
// D# group0: count=1 | lds_addr[63:32] | global_addr[120:64] | type=2[127:126]
// D# group1: data_size=2B | tensor_dim0/1 | tile_dim0/1 | dim0_stride
__device__ __forceinline__ void tdm_load_2d(unsigned lds_addr, const void* gptr,
                                            unsigned tile_w, unsigned tile_h,
                                            unsigned stride_elems) {
    const unsigned long long ga = (unsigned long long)gptr;
    u32x4 g0;
    g0[0] = __builtin_amdgcn_readfirstlane(1u);
    g0[1] = __builtin_amdgcn_readfirstlane(lds_addr);
    g0[2] = __builtin_amdgcn_readfirstlane((unsigned)ga);
    g0[3] = __builtin_amdgcn_readfirstlane(((unsigned)(ga >> 32) & 0x01FFFFFFu) | 0x80000000u);
    u32x8 g1;
    g1[0] = 0x00010000u;                // workgroup_mask=0, data_size=1 (2 bytes)
    g1[1] = tile_w << 16;               // tensor_dim0[15:0] (atomic barrier addr=0)
    g1[2] = tile_h << 16;               // tensor_dim1[15:0] (tensor_dim0 hi=0)
    g1[3] = tile_w << 16;               // tile_dim0 (tensor_dim1 hi=0)
    g1[4] = tile_h;                     // tile_dim1 (tile_dim2=0)
    g1[5] = stride_elems;               // tensor_dim0_stride[31:0]
    g1[6] = 0u;
    g1[7] = 0u;
    asm volatile("tensor_load_to_lds %0, %1" :: "s"(g0), "s"(g1) : "memory");
}

union Frag { v16h v; h8 h[2]; };

// ---------------------------------------------------------------------------
// GEMM1: S_k = Gh_k[1024x1024] @ CombX^T  (CombX stored [col=b*65+p][j])
// 256(M) x 64(cols) tile, K-step 32, double-buffered async-to-LDS staging.
// 8 waves; each wave owns 32 rows = 2 A-frags x 4 B-frags = 8 wmma/K-step
// (B-frags reused across both A-frags). Fragments preloaded so ds_loads batch
// and the 8 wmmas issue back-to-back. Output scattered f16 to scf16[b][n][..].
// ---------------------------------------------------------------------------
__global__ __launch_bounds__(256) void gemm_gcn(const _Float16* __restrict__ Gh,
                                                const _Float16* __restrict__ CombX,
                                                _Float16* __restrict__ scf16) {
    __shared__ __align__(16) _Float16 As[2][256 * 32];   // [row][k]
    __shared__ __align__(16) _Float16 Bs[2][64 * 32];    // [col][k] K-contiguous
    const int tid  = threadIdx.x;
    const int wave = tid >> 5, lane = tid & 31;
    const int mbase = blockIdx.x * 256;
    const int cbase = blockIdx.y * 64;
    const int kk    = blockIdx.z;
    const _Float16* A = Gh + (size_t)kk * NN * NN;
    const unsigned asb = lds_lo32(&As[0][0]);
    const unsigned bsb = lds_lo32(&Bs[0][0]);

    const int rA0 = tid >> 2, kc = (tid & 3) * 8;  // 16B chunks: 4 per 32-elem row

    v8f acc[2][4] = {};
    const int lan15  = lane & 15;
    const int khalfA = (lane >> 4) * 16;
    const int kbB    = (lane >> 4) * 16;

#define FILL_G1(buf, k0)                                                          \
    do {                                                                          \
        const unsigned ab = asb + (buf) * (256 * 32 * 2);                         \
        const unsigned bb = bsb + (buf) * (64 * 32 * 2);                          \
        async_ld16(ab + ((rA0      ) * 32 + kc) * 2, &A[(size_t)(mbase + rA0      ) * NN + (k0) + kc]); \
        async_ld16(ab + ((rA0 +  64) * 32 + kc) * 2, &A[(size_t)(mbase + rA0 +  64) * NN + (k0) + kc]); \
        async_ld16(ab + ((rA0 + 128) * 32 + kc) * 2, &A[(size_t)(mbase + rA0 + 128) * NN + (k0) + kc]); \
        async_ld16(ab + ((rA0 + 192) * 32 + kc) * 2, &A[(size_t)(mbase + rA0 + 192) * NN + (k0) + kc]); \
        async_ld16(bb + (rA0 * 32 + kc) * 2, &CombX[(size_t)(cbase + rA0) * NN + (k0) + kc]);           \
    } while (0)

    FILL_G1(0, 0);
    int cur = 0;
    for (int k0 = 0; k0 < NN; k0 += 32) {
        wait_async0();              // my async fills of buf[cur] complete
        __syncthreads();            // everyone's fills done; prev reads of buf[cur^1] done
        if (k0 + 32 < NN) FILL_G1(cur ^ 1, k0 + 32);
        const _Float16* as = &As[cur][0];
        const _Float16* bs = &Bs[cur][0];
        Frag ua0, ua1, ub[4];
        const int ar0 = wave * 32 + lan15, ar1 = ar0 + 16;
        ua0.h[0] = *(const h8*)&as[ar0 * 32 + khalfA];
        ua0.h[1] = *(const h8*)&as[ar0 * 32 + khalfA + 8];
        ua1.h[0] = *(const h8*)&as[ar1 * 32 + khalfA];
        ua1.h[1] = *(const h8*)&as[ar1 * 32 + khalfA + 8];
        #pragma unroll
        for (int jt = 0; jt < 4; ++jt) {
            const int bcol = jt * 16 + lan15;
            ub[jt].h[0] = *(const h8*)&bs[bcol * 32 + kbB];
            ub[jt].h[1] = *(const h8*)&bs[bcol * 32 + kbB + 8];
        }
        #pragma unroll
        for (int jt = 0; jt < 4; ++jt) {
            acc[0][jt] = __builtin_amdgcn_wmma_f32_16x16x32_f16(
                false, ua0.v, false, ub[jt].v, (short)0, acc[0][jt], false, false);
            acc[1][jt] = __builtin_amdgcn_wmma_f32_16x16x32_f16(
                false, ua1.v, false, ub[jt].v, (short)0, acc[1][jt], false, false);
        }
        cur ^= 1;
    }
#undef FILL_G1
    const int hi8 = (lane >> 4) * 8;
    #pragma unroll
    for (int jt = 0; jt < 4; ++jt) {
        const int cg = cbase + jt * 16 + lan15;   // global column = b*65+p
        const int b = cg / 65, p = cg - b * 65;
        const size_t base = (size_t)b * NN * PPAD + kk * 65 + p;
        #pragma unroll
        for (int mi = 0; mi < 2; ++mi) {
            #pragma unroll
            for (int v = 0; v < 8; ++v) {
                const int n = mbase + wave * 32 + mi * 16 + v + hi8;
                scf16[base + (size_t)n * PPAD] = (_Float16)acc[mi][jt][v];
            }
        }
    }
}

// ---------------------------------------------------------------------------
// GEMM2: conv[b] = scf16[b][1024x224] @ WT^T + bias.  A-tile: async-to-LDS.
// B (weight) tile: one TDM 2-D descriptor per K-step, issued by wave 0,
// synchronized via TENSORcnt. Double-buffered; fragments preloaded.
// ---------------------------------------------------------------------------
__global__ __launch_bounds__(256) void gemm_conv(const _Float16* __restrict__ scf16,
                                                 const _Float16* __restrict__ WT,
                                                 long long wt_stride,
                                                 const float* __restrict__ bias,
                                                 int bias_stride,
                                                 float* __restrict__ conv) {
    __shared__ __align__(16) _Float16 As[2][128 * 32];
    __shared__ __align__(16) _Float16 Bs[2][128 * 32];
    const int tid  = threadIdx.x;
    const int wave = tid >> 5, lane = tid & 31;
    const int mbase = blockIdx.x * 128;
    const int qbase = blockIdx.y * 128;
    const int b     = blockIdx.z;
    const _Float16* A  = scf16 + (size_t)b * NN * PPAD;
    const _Float16* Bm = WT + (size_t)b * wt_stride;
    const unsigned asb = lds_lo32(&As[0][0]);
    const unsigned bsb = lds_lo32(&Bs[0][0]);

    const int rA0 = tid >> 2, kc = (tid & 3) * 8;
    const int rA1 = rA0 + 64;

    v8f acc[8] = {};
    const int lan15  = lane & 15;
    const int khalfA = (lane >> 4) * 16;
    const int kbB    = (lane >> 4) * 16;

#define FILL_A2(buf, k0)                                                          \
    do {                                                                          \
        const unsigned ab = asb + (buf) * (128 * 32 * 2);                         \
        async_ld16(ab + (rA0 * 32 + kc) * 2, &A[(size_t)(mbase + rA0) * PPAD + (k0) + kc]); \
        async_ld16(ab + (rA1 * 32 + kc) * 2, &A[(size_t)(mbase + rA1) * PPAD + (k0) + kc]); \
    } while (0)
#define FILL_B2(buf, k0)                                                          \
    do {                                                                          \
        if (wave == 0)                                                            \
            tdm_load_2d(bsb + (buf) * (128 * 32 * 2),                             \
                        &Bm[(size_t)qbase * PPAD + (k0)], 32u, 128u, PPAD);       \
    } while (0)

    FILL_A2(0, 0);
    FILL_B2(0, 0);
    int cur = 0;
    for (int k0 = 0; k0 < PPAD; k0 += 32) {
        wait_async0();
        if (wave == 0) __builtin_amdgcn_s_wait_tensorcnt(0);
        __syncthreads();
        if (k0 + 32 < PPAD) { FILL_A2(cur ^ 1, k0 + 32); FILL_B2(cur ^ 1, k0 + 32); }
        const _Float16* as = &As[cur][0];
        const _Float16* bs = &Bs[cur][0];
        Frag ua, ub[8];
        const int arow = wave * 16 + lan15;
        ua.h[0] = *(const h8*)&as[arow * 32 + khalfA];
        ua.h[1] = *(const h8*)&as[arow * 32 + khalfA + 8];
        #pragma unroll
        for (int jt = 0; jt < 8; ++jt) {
            const int bcol = jt * 16 + lan15;
            ub[jt].h[0] = *(const h8*)&bs[bcol * 32 + kbB];
            ub[jt].h[1] = *(const h8*)&bs[bcol * 32 + kbB + 8];
        }
        #pragma unroll
        for (int jt = 0; jt < 8; ++jt) {
            acc[jt] = __builtin_amdgcn_wmma_f32_16x16x32_f16(
                false, ua.v, false, ub[jt].v, (short)0, acc[jt], false, false);
        }
        cur ^= 1;
    }
#undef FILL_A2
#undef FILL_B2
    const float* bp = bias + (size_t)b * bias_stride;
    const int hi8 = (lane >> 4) * 8;
    #pragma unroll
    for (int jt = 0; jt < 8; ++jt) {
        const int q = qbase + jt * 16 + lan15;
        const float bv = bp[q];
        #pragma unroll
        for (int v = 0; v < 8; ++v) {
            const int n = mbase + wave * 16 + v + hi8;
            conv[((size_t)(b << 10) + n) * NG4 + q] = acc[jt][v] + bv;
        }
    }
}

// ---------------------------------------------------------------------------
// LSTM gate update (one thread per (b,n)). Writes c state, repacks h (f16)
// into CombX h-slots; in decoder mode also projects h and feeds output back.
// ---------------------------------------------------------------------------
__global__ __launch_bounds__(256) void lstm_update(const float* __restrict__ conv,
                                                   float* __restrict__ cst,
                                                   _Float16* __restrict__ CombX,
                                                   const float* __restrict__ proj_w,
                                                   const float* __restrict__ proj_b,
                                                   float* __restrict__ out, int step) {
    const int gid = blockIdx.x * 256 + threadIdx.x;   // b*1024 + n
    const int b = gid >> 10, n = gid & 1023;
    const float* cv = conv + (size_t)gid * NG4;
    float pacc = 0.f;
    for (int q = 0; q < NH; ++q) {
        const float gi = cv[q], gf = cv[NH + q], go = cv[2 * NH + q], gg = cv[3 * NH + q];
        const float i = 1.f / (1.f + __expf(-gi));
        const float f = 1.f / (1.f + __expf(-gf));
        const float o = 1.f / (1.f + __expf(-go));
        const float g = tanhf(gg);
        const float c = f * cst[(size_t)gid * NH + q] + i * g;
        cst[(size_t)gid * NH + q] = c;
        const float h = o * tanhf(c);
        CombX[(size_t)(b * NDIN + 1 + q) * NN + n] = (_Float16)h;
        if (out) pacc += h * proj_w[q];
    }
    if (out) {
        const float val = pacc + proj_b[0];
        out[((size_t)b * NT + step) * NN + n] = val;            // [B,HOR,N,1]
        CombX[(size_t)(b * NDIN) * NN + n] = (_Float16)val;     // autoregressive feed
    }
}

// ---------------------------------------------------------------------------
// Small helper kernels
// ---------------------------------------------------------------------------
__global__ void pack_x(const float* __restrict__ x_seq, _Float16* __restrict__ CombX, int t) {
    const int gid = blockIdx.x * 256 + threadIdx.x;
    const int b = gid >> 10, n = gid & 1023;
    CombX[(size_t)(b * NDIN) * NN + n] =
        (_Float16)x_seq[((size_t)b * NT + t) * NN + n];
}

__global__ void zero_xslot(_Float16* __restrict__ CombX) {
    const int gid = blockIdx.x * 256 + threadIdx.x;
    const int b = gid >> 10, n = gid & 1023;
    CombX[(size_t)(b * NDIN) * NN + n] = (_Float16)0.f;
}

__global__ void meta_hidden(const float* __restrict__ x_meta,
                            const float* __restrict__ lw1_w, const float* __restrict__ lw1_b,
                            const float* __restrict__ lb1_w, const float* __restrict__ lb1_b,
                            float* __restrict__ hmid, float* __restrict__ hbias, int t) {
    const int b = blockIdx.x, tid = threadIdx.x;  // 128 threads: 0-63 W-path, 64-127 b-path
    const float* m = x_meta + ((size_t)b * NT + t) * 32;
    const int i = tid & 63;
    const float* W  = (tid < 64) ? lw1_w : lb1_w;
    const float* bb = (tid < 64) ? lw1_b : lb1_b;
    float s = bb[i];
    for (int j = 0; j < 32; ++j) s += m[j] * W[j * 64 + i];
    s = fmaxf(s, 0.f);
    ((tid < 64) ? hmid : hbias)[b * 64 + i] = s;
}

// WmT[b][q][p] = lw2_b[p*256+q] + hmid[b] . lw2_w[:, p*256+q]  (p padded to 224)
__global__ void meta_w(const float* __restrict__ hmid, const float* __restrict__ lw2_w,
                       const float* __restrict__ lw2_b, _Float16* __restrict__ WmT) {
    const int p = blockIdx.x, b = blockIdx.y, q = threadIdx.x;
    _Float16* dst = WmT + ((size_t)b * NG4 + q) * PPAD + p;
    if (p >= 195) { *dst = (_Float16)0.f; return; }
    const float* hm = hmid + b * 64;
    float s = lw2_b[p * NG4 + q];
    for (int i = 0; i < 64; ++i) s += hm[i] * lw2_w[(size_t)i * 49920 + p * NG4 + q];
    *dst = (_Float16)s;
}

__global__ void meta_b(const float* __restrict__ hbias, const float* __restrict__ lb2_w,
                       const float* __restrict__ lb2_b, float* __restrict__ bm) {
    const int b = blockIdx.x, q = threadIdx.x;
    const float* hb = hbias + b * 64;
    float s = lb2_b[q];
    for (int i = 0; i < 64; ++i) s += hb[i] * lb2_w[i * NG4 + q];
    bm[b * NG4 + q] = s;
}

__global__ void cvt_half(const float* __restrict__ src, _Float16* __restrict__ dst, size_t n) {
    size_t i = (size_t)blockIdx.x * blockDim.x + threadIdx.x;
    const size_t stride = (size_t)gridDim.x * blockDim.x;
    for (; i < n; i += stride) dst[i] = (_Float16)src[i];
}

__global__ void pack_decw(const float* __restrict__ dec_W, _Float16* __restrict__ decWT) {
    const int p = blockIdx.x, q = threadIdx.x;
    decWT[(size_t)q * PPAD + p] = (p < 195) ? (_Float16)dec_W[p * NG4 + q] : (_Float16)0.f;
}

__global__ void zero_mem(unsigned* __restrict__ p, size_t nwords) {
    size_t i = (size_t)blockIdx.x * blockDim.x + threadIdx.x;
    const size_t stride = (size_t)gridDim.x * blockDim.x;
    for (; i < nwords; i += stride) p[i] = 0u;
}

// ---------------------------------------------------------------------------
extern "C" void kernel_launch(void* const* d_in, const int* in_sizes, int n_in,
                              void* d_out, int out_size, void* d_ws, size_t ws_size,
                              hipStream_t stream) {
    (void)in_sizes; (void)n_in; (void)out_size; (void)ws_size;
    const float* G      = (const float*)d_in[0];
    const float* x_seq  = (const float*)d_in[1];
    const float* x_meta = (const float*)d_in[2];
    const float* lw1_w  = (const float*)d_in[3];
    const float* lw1_b  = (const float*)d_in[4];
    const float* lw2_w  = (const float*)d_in[5];
    const float* lw2_b  = (const float*)d_in[6];
    const float* lb1_w  = (const float*)d_in[7];
    const float* lb1_b  = (const float*)d_in[8];
    const float* lb2_w  = (const float*)d_in[9];
    const float* lb2_b  = (const float*)d_in[10];
    const float* dec_W  = (const float*)d_in[11];
    const float* dec_b  = (const float*)d_in[12];
    const float* proj_w = (const float*)d_in[13];
    const float* proj_b = (const float*)d_in[14];
    float* out = (float*)d_out;

    // ---- workspace carve-up (all offsets 256B aligned) ----
    char* ws = (char*)d_ws;
    size_t off = 0;
    auto carve = [&](size_t bytes) {
        void* p = ws + off;
        off += (bytes + 255) & ~(size_t)255;
        return p;
    };
    _Float16* Gh     = (_Float16*)carve((size_t)3 * NN * NN * 2);        //  6.0 MB
    _Float16* CombX  = (_Float16*)carve((size_t)NCOL * NN * 2);          //  8.5 MB
    _Float16* scf16  = (_Float16*)carve((size_t)NB * NN * PPAD * 2);     // 29.4 MB
    _Float16* WmT    = (_Float16*)carve((size_t)NB * NG4 * PPAD * 2);    //  7.3 MB
    _Float16* decWT  = (_Float16*)carve((size_t)NG4 * PPAD * 2);         //  0.1 MB
    float*    bm     = (float*)carve((size_t)NB * NG4 * 4);
    float*    hmid   = (float*)carve((size_t)NB * 64 * 4);
    float*    hbias  = (float*)carve((size_t)NB * 64 * 4);
    float*    conv   = (float*)carve((size_t)NB * NN * NG4 * 4);         // 64.0 MB
    float*    cst    = (float*)carve((size_t)NB * NN * NH * 4);          // 16.8 MB
    const size_t total = off;

    // deterministic init: zero everything we use (h0=c0=0, f16 pad lanes = 0)
    zero_mem<<<4096, 256, 0, stream>>>((unsigned*)d_ws, total / 4);
    cvt_half<<<4096, 256, 0, stream>>>(G, Gh, (size_t)3 * NN * NN);
    pack_decw<<<PPAD, NG4, 0, stream>>>(dec_W, decWT);

    const dim3 gG(4, NCOL / 64, 3);     // GEMM1: 4 M-tiles x 65 col-tiles x 3 supports
    const dim3 gC(8, NG4 / 128, NB);    // GEMM2: 8 M-tiles x 2 q-tiles x 64 batches

    // ---------------- Encoder ----------------
    for (int t = 0; t < NT; ++t) {
        pack_x<<<NB * NN / 256, 256, 0, stream>>>(x_seq, CombX, t);
        meta_hidden<<<NB, 128, 0, stream>>>(x_meta, lw1_w, lw1_b, lb1_w, lb1_b,
                                            hmid, hbias, t);
        meta_w<<<dim3(PPAD, NB), NG4, 0, stream>>>(hmid, lw2_w, lw2_b, WmT);
        meta_b<<<NB, NG4, 0, stream>>>(hbias, lb2_w, lb2_b, bm);
        gemm_gcn<<<gG, 256, 0, stream>>>(Gh, CombX, scf16);
        gemm_conv<<<gC, 256, 0, stream>>>(scf16, WmT, (long long)NG4 * PPAD,
                                          bm, NG4, conv);
        lstm_update<<<NB * NN / 256, 256, 0, stream>>>(conv, cst, CombX,
                                                       proj_w, proj_b, nullptr, 0);
    }

    // ---------------- Decoder ----------------
    zero_xslot<<<NB * NN / 256, 256, 0, stream>>>(CombX);   // first decoder input = 0
    for (int s = 0; s < NT; ++s) {
        gemm_gcn<<<gG, 256, 0, stream>>>(Gh, CombX, scf16);
        gemm_conv<<<gC, 256, 0, stream>>>(scf16, decWT, 0, dec_b, 0, conv);
        lstm_update<<<NB * NN / 256, 256, 0, stream>>>(conv, cst, CombX,
                                                       proj_w, proj_b, out, s);
    }
}
